// GraphAttentionLayer_30726196036134
// MI455X (gfx1250) — compile-verified, matrix operand order checked by reading, not run on hardware
//
#include <hip/hip_runtime.h>
#include <hip/hip_bf16.h>
#include <math.h>

typedef __attribute__((ext_vector_type(16))) _Float16 v16h;
typedef __attribute__((ext_vector_type(2)))  _Float16 v2h;
typedef __attribute__((ext_vector_type(8)))  float    v8f;

#define IN_F   128
#define OUT_F  64
#define DEG    16
#define ALPHA  0.2f
#define LDSK   132   // padded K-stride for LDS weight tiles (even -> 4B aligned pairs)

// K index inside a 16-bit 16x32 A/B fragment for VGPR-pair v (0..7), lane-half h (0/1).
// CDNA5 ISA: lanes 0-15 VGPR0-3 hold K=0..7, VGPR4-7 hold K=16..23; lanes 16-31 add +8.
__device__ __forceinline__ int wmma_k(int v, int h) {
    return ((v < 4) ? (v * 2) : (16 + (v - 4) * 2)) + h * 8;
}

// ---------------------------------------------------------------------------
// Kernel 1: h_high = f32(input) @ W_high, h_low = input @ W_low  (f16 WMMA,
// f32 accumulate). 256 threads = 8 waves; wave w owns rows [base+16w, +16).
// ---------------------------------------------------------------------------
__global__ __launch_bounds__(256)
void gemm_wmma_kernel(const float* __restrict__ input,
                      const float* __restrict__ W_high,
                      const float* __restrict__ W_low,
                      float* __restrict__ h_high,
                      float* __restrict__ h_low,
                      int n)
{
    __shared__ _Float16 ldsW[2][OUT_F * LDSK];   // transposed [n][k], padded

    // Stage both weight matrices into LDS as f16 (coalesced global reads).
    for (int idx = threadIdx.x; idx < IN_F * OUT_F; idx += 256) {
        int k = idx >> 6;          // 0..127
        int c = idx & 63;          // 0..63
        ldsW[0][c * LDSK + k] = (_Float16)W_high[idx];
        ldsW[1][c * LDSK + k] = (_Float16)W_low[idx];
    }
    __syncthreads();

    const int wave = threadIdx.x >> 5;
    const int lane = threadIdx.x & 31;
    const int half = lane >> 4;                 // K-half select
    const int rowbase = blockIdx.x * 128 + wave * 16;

    // ---- Build the 4 A fragments (K = 0..127) once. Clamp OOB rows (stores
    // are guarded later); keeps EXEC all-ones as WMMA requires. ----
    int m = rowbase + (lane & 15);
    if (m >= n) m = n - 1;
    const float* arow = input + (size_t)m * IN_F;

    v16h afrag[4];
#pragma unroll
    for (int kf = 0; kf < 4; ++kf) {
#pragma unroll
        for (int v = 0; v < 8; ++v) {
            int k0 = kf * 32 + wmma_k(v, half);          // even
            float2 f = *(const float2*)(arow + k0);      // 8B aligned
            afrag[kf][2 * v]     = (_Float16)f.x;
            afrag[kf][2 * v + 1] = (_Float16)f.y;
        }
    }

    const int nlocal = lane & 15;
    const int madd   = half * 8;

#pragma unroll
    for (int nt = 0; nt < 4; ++nt) {                     // output col tile
        v8f accH = {};
        v8f accL = {};
        const _Float16* bcolH = &ldsW[0][(nt * 16 + nlocal) * LDSK];
        const _Float16* bcolL = &ldsW[1][(nt * 16 + nlocal) * LDSK];
#pragma unroll
        for (int kf = 0; kf < 4; ++kf) {
            v16h bH, bL;
#pragma unroll
            for (int v = 0; v < 8; ++v) {
                int k0 = kf * 32 + wmma_k(v, half);      // even -> 4B aligned
                v2h pH = *(const v2h*)(bcolH + k0);
                v2h pL = *(const v2h*)(bcolL + k0);
                bH[2 * v] = pH.x; bH[2 * v + 1] = pH.y;
                bL[2 * v] = pL.x; bL[2 * v + 1] = pL.y;
            }
            accH = __builtin_amdgcn_wmma_f32_16x16x32_f16(
                false, afrag[kf], false, bH, (short)0, accH, false, false);
            accL = __builtin_amdgcn_wmma_f32_16x16x32_f16(
                false, afrag[kf], false, bL, (short)0, accL, false, false);
        }
        // D layout: lane half*? -> rows r+8*half, col = nt*16 + nlocal
#pragma unroll
        for (int r = 0; r < 8; ++r) {
            int row = rowbase + r + madd;
            if (row < n) {
                size_t o = (size_t)row * OUT_F + nt * 16 + nlocal;
                h_high[o] = accH[r];
                h_low[o]  = accL[r];
            }
        }
    }
}

// ---------------------------------------------------------------------------
// Kernel 2: per-node attention scalars  g = h . a_half   (wave per node)
// ---------------------------------------------------------------------------
__global__ __launch_bounds__(256)
void attn_score_kernel(const float* __restrict__ h_high,
                       const float* __restrict__ h_low,
                       const float* __restrict__ a_high,
                       const float* __restrict__ a_low,
                       float* __restrict__ gsH, float* __restrict__ gdH,
                       float* __restrict__ gsL, float* __restrict__ gdL,
                       int n)
{
    int node = (int)((blockIdx.x * blockDim.x + threadIdx.x) >> 5);
    int lane = threadIdx.x & 31;
    if (node >= n) return;

    const float* hh = h_high + (size_t)node * OUT_F;
    const float* hl = h_low  + (size_t)node * OUT_F;
    float x0 = hh[lane], x1 = hh[lane + 32];
    float y0 = hl[lane], y1 = hl[lane + 32];

    float sH = x0 * a_high[lane]      + x1 * a_high[lane + 32];
    float dH = x0 * a_high[64 + lane] + x1 * a_high[96 + lane];
    float sL = y0 * a_low[lane]       + y1 * a_low[lane + 32];
    float dL = y0 * a_low[64 + lane]  + y1 * a_low[96 + lane];

#pragma unroll
    for (int msk = 16; msk >= 1; msk >>= 1) {
        sH += __shfl_xor(sH, msk, 32);
        dH += __shfl_xor(dH, msk, 32);
        sL += __shfl_xor(sL, msk, 32);
        dL += __shfl_xor(dL, msk, 32);
    }
    if (lane == 0) {
        gsH[node] = sH; gdH[node] = dH;
        gsL[node] = sL; gdL[node] = dL;
    }
}

// ---------------------------------------------------------------------------
// Kernel 3: per-node edge weights + rowsums + h_agg  (wave per node, DEG=16)
// ---------------------------------------------------------------------------
__global__ __launch_bounds__(256)
void edge_agg_kernel(const int* __restrict__ dstIdx,
                     const float* __restrict__ h_high,
                     const float* __restrict__ h_low,
                     const float* __restrict__ gsH, const float* __restrict__ gdH,
                     const float* __restrict__ gsL, const float* __restrict__ gdL,
                     float* __restrict__ ehH, float* __restrict__ ehL,
                     float* __restrict__ rowsumH, float* __restrict__ rowsumL,
                     float* __restrict__ aggH, float* __restrict__ aggL,
                     int n)
{
    int node = (int)((blockIdx.x * blockDim.x + threadIdx.x) >> 5);
    int lane = threadIdx.x & 31;
    if (node >= n) return;

    const int ebase = node * DEG;
    const int e = lane & 15;                   // lanes 16-31 mirror 0-15
    const int d = dstIdx[ebase + e];

    float sH = gsH[node] + gdH[d];
    float sL = gsL[node] + gdL[d];
    float eHraw = __expf(-(sH > 0.f ? sH : ALPHA * sH));
    float eLraw = __expf(-(sL > 0.f ? sL : ALPHA * sL));

    if (lane < 16) {                           // relu6 (values are > 0)
        ehH[ebase + e] = fminf(eHraw, 6.f);
        ehL[ebase + e] = fminf(eLraw, 6.f);
    }

    // rowsum over the 16 edges uses the RAW exp values (pre-relu6).
    float rH = eHraw, rL = eLraw;
#pragma unroll
    for (int msk = 8; msk >= 1; msk >>= 1) {   // reduce within each 16-lane half
        rH += __shfl_xor(rH, msk, 16);
        rL += __shfl_xor(rL, msk, 16);
    }
    if (lane == 0) { rowsumH[node] = rH; rowsumL[node] = rL; }

    // h_agg_high[i] = 16*h_high[i] + sum_e h_high[dst]
    // h_agg_low [i] = 16*h_low [i] - sum_e h_low [dst]
    const size_t nb = (size_t)node * OUT_F;
    float aH0 = 16.f * h_high[nb + lane], aH1 = 16.f * h_high[nb + lane + 32];
    float aL0 = 16.f * h_low [nb + lane], aL1 = 16.f * h_low [nb + lane + 32];
#pragma unroll
    for (int j = 0; j < DEG; ++j) {
        int dj = __shfl(d, j, 32);
        const float* hh = h_high + (size_t)dj * OUT_F;
        const float* hl = h_low  + (size_t)dj * OUT_F;
        aH0 += hh[lane]; aH1 += hh[lane + 32];
        aL0 -= hl[lane]; aL1 -= hl[lane + 32];
    }
    aggH[nb + lane] = aH0; aggH[nb + lane + 32] = aH1;
    aggL[nb + lane] = aL0; aggL[nb + lane + 32] = aL1;
}

// ---------------------------------------------------------------------------
// Kernel 4: SpMM + normalize + mix + relu6  (wave per node)
// ---------------------------------------------------------------------------
__global__ __launch_bounds__(256)
void spmm_out_kernel(const int* __restrict__ dstIdx,
                     const float* __restrict__ ehH, const float* __restrict__ ehL,
                     const float* __restrict__ rowsumH, const float* __restrict__ rowsumL,
                     const float* __restrict__ aggH, const float* __restrict__ aggL,
                     float* __restrict__ out, int n)
{
    int node = (int)((blockIdx.x * blockDim.x + threadIdx.x) >> 5);
    int lane = threadIdx.x & 31;
    if (node >= n) return;

    const int ebase = node * DEG;
    const int e = lane & 15;
    const int d  = dstIdx[ebase + e];
    const float wH = ehH[ebase + e];
    const float wL = ehL[ebase + e];

    float oH0 = 0.f, oH1 = 0.f, oL0 = 0.f, oL1 = 0.f;
#pragma unroll
    for (int j = 0; j < DEG; ++j) {
        int   dj = __shfl(d,  j, 32);
        float wh = __shfl(wH, j, 32);
        float wl = __shfl(wL, j, 32);
        const float* ah = aggH + (size_t)dj * OUT_F;
        const float* al = aggL + (size_t)dj * OUT_F;
        oH0 = fmaf(wh, ah[lane],      oH0);
        oH1 = fmaf(wh, ah[lane + 32], oH1);
        oL0 = fmaf(wl, al[lane],      oL0);
        oL1 = fmaf(wl, al[lane + 32], oL1);
    }
    float invH = 1.f / rowsumH[node];
    float invL = 1.f / rowsumL[node];
    float o0 = 0.5f * (oH0 * invH + oL0 * invL);
    float o1 = 0.5f * (oH1 * invH + oL1 * invL);
    size_t nb = (size_t)node * OUT_F;
    out[nb + lane]      = fminf(fmaxf(o0, 0.f), 6.f);
    out[nb + lane + 32] = fminf(fmaxf(o1, 0.f), 6.f);
}

// ---------------------------------------------------------------------------
extern "C" void kernel_launch(void* const* d_in, const int* in_sizes, int n_in,
                              void* d_out, int out_size, void* d_ws, size_t ws_size,
                              hipStream_t stream)
{
    const float* input  = (const float*)d_in[0];
    const int*   edge   = (const int*)  d_in[1];
    const float* W_high = (const float*)d_in[2];
    const float* W_low  = (const float*)d_in[3];
    const float* a_high = (const float*)d_in[4];
    const float* a_low  = (const float*)d_in[5];
    float* out = (float*)d_out;

    const int N = in_sizes[0] / IN_F;        // 100000
    const int E = in_sizes[1] / 2;           // N * DEG
    const int* dstIdx = edge + E;            // edge[1] row

    // Workspace carve-up (256B aligned chunks)
    auto align = [](size_t x) { return (x + 255) & ~(size_t)255; };
    char* p = (char*)d_ws;
    float* h_high  = (float*)p; p += align((size_t)N * OUT_F * 4);
    float* h_low   = (float*)p; p += align((size_t)N * OUT_F * 4);
    float* aggH    = (float*)p; p += align((size_t)N * OUT_F * 4);
    float* aggL    = (float*)p; p += align((size_t)N * OUT_F * 4);
    float* ehH     = (float*)p; p += align((size_t)E * 4);
    float* ehL     = (float*)p; p += align((size_t)E * 4);
    float* gsH     = (float*)p; p += align((size_t)N * 4);
    float* gdH     = (float*)p; p += align((size_t)N * 4);
    float* gsL     = (float*)p; p += align((size_t)N * 4);
    float* gdL     = (float*)p; p += align((size_t)N * 4);
    float* rowsumH = (float*)p; p += align((size_t)N * 4);
    float* rowsumL = (float*)p; p += align((size_t)N * 4);

    // 1) GEMMs via f16 WMMA (8 waves x 16 rows = 128 rows / block)
    int gemmBlocks = (N + 127) / 128;
    gemm_wmma_kernel<<<gemmBlocks, 256, 0, stream>>>(
        input, W_high, W_low, h_high, h_low, N);

    // 2) per-node attention scalars (wave per node)
    int nodeBlocks = (N + 7) / 8;
    attn_score_kernel<<<nodeBlocks, 256, 0, stream>>>(
        h_high, h_low, a_high, a_low, gsH, gdH, gsL, gdL, N);

    // 3) edge weights, rowsums, h_agg
    edge_agg_kernel<<<nodeBlocks, 256, 0, stream>>>(
        dstIdx, h_high, h_low, gsH, gdH, gsL, gdL,
        ehH, ehL, rowsumH, rowsumL, aggH, aggL, N);

    // 4) SpMM + normalize + relu6
    spmm_out_kernel<<<nodeBlocks, 256, 0, stream>>>(
        dstIdx, ehH, ehL, rowsumH, rowsumL, aggH, aggL, out, N);
}
